// resSCNN_sad_26491358282342
// MI455X (gfx1250) — compile-verified, matrix operand order checked by reading, not run on hardware
//
// MI455X (gfx1250) implementation of the resSCNN forward pass.
// Core: implicit-GEMM conv via v_wmma_f32_16x16x32_f16, wave32.
//  - Block = 256 threads = 8 waves; block tile = 128 couts x 16 pixels.
//  - B (im2col) panel in LDS, column-major stride-72 f16 -> b128 ds reads,
//    double buffered, K-step 64 => two back-to-back WMMAs per iteration.
//  - A (weights) pre-converted to f16 with K padded to 64, streamed b128.
//  - BN = two-pass batch stats (atomics) + fused scale/shift/ReLU/residual.
//  - TDM probe kernel (tensor_load_to_lds + s_wait_tensorcnt), 6-arg form.
#include <hip/hip_runtime.h>

typedef __attribute__((ext_vector_type(16))) _Float16 v16h;
typedef __attribute__((ext_vector_type(8)))  _Float16 v8h;
typedef __attribute__((ext_vector_type(8)))  float    v8f;
typedef __attribute__((ext_vector_type(4)))  unsigned int v4u;
typedef __attribute__((ext_vector_type(8)))  int      v8i;
typedef __attribute__((ext_vector_type(4)))  int      v4i;

#define DEVINL __device__ __forceinline__

// ---------------- elementwise / conversion kernels ----------------

__global__ void k_cvt_w(const float* __restrict__ w, _Float16* __restrict__ o,
                        int Cout, int K, int Kp) {
  long i = (long)blockIdx.x * blockDim.x + threadIdx.x;
  long total = (long)Cout * Kp;
  if (i >= total) return;
  int m = (int)(i / Kp), k = (int)(i % Kp);
  o[i] = (k < K) ? (_Float16)w[(long)m * K + k] : (_Float16)0.f;
}

__global__ void k_cvt16(const float* __restrict__ x, _Float16* __restrict__ o, long n) {
  long i = (long)blockIdx.x * blockDim.x + threadIdx.x;
  if (i < n) o[i] = (_Float16)x[i];
}

__global__ void k_zero(float* __restrict__ p, long n) {
  long i = (long)blockIdx.x * blockDim.x + threadIdx.x;
  if (i < n) p[i] = 0.f;
}

__global__ void k_relu_cvt(float* __restrict__ x, _Float16* __restrict__ o16, long n) {
  long i = (long)blockIdx.x * blockDim.x + threadIdx.x;
  if (i >= n) return;
  float v = fmaxf(x[i], 0.f);
  x[i] = v;
  o16[i] = (_Float16)v;
}

__global__ void k_add_cvt(float* __restrict__ a, const float* __restrict__ b,
                          _Float16* __restrict__ o16, long n) {
  long i = (long)blockIdx.x * blockDim.x + threadIdx.x;
  if (i >= n) return;
  float v = a[i] + b[i];
  a[i] = v;
  o16[i] = (_Float16)v;
}

__global__ void k_sigmoid(const float* __restrict__ x, float* __restrict__ o, long n) {
  long i = (long)blockIdx.x * blockDim.x + threadIdx.x;
  if (i < n) o[i] = 1.f / (1.f + expf(-x[i]));
}

// ---------------- TDM probe: stage a 16x64 f16 tile via the Tensor Data Mover ----
// Descriptor packing per CDNA5 ISA (group0: count,
// lds_addr, global_addr, type=2; group1: data_size, tensor/tile dims).
// This toolchain exposes the 6-arg builtin:
//   (uint32x4 g0, int32x8 g1, int32x4 g2, int32x4 g3, int32x8 g4, i32 cpol)
#if __has_builtin(__builtin_amdgcn_tensor_load_to_lds)
__global__ void k_tdm_stage(const _Float16* __restrict__ src, _Float16* __restrict__ dst,
                            int tensor_w, int tensor_h, int row_stride) {
  __shared__ _Float16 tile[16 * 64];
  unsigned long long ga = (unsigned long long)(const void*)src;
  unsigned int lds_off = (unsigned int)(unsigned long long)(void*)&tile[0];
  v4u g0;
  g0.x = 1u;                                      // count=1, is_restore=0
  g0.y = lds_off;                                 // lds_addr
  g0.z = (unsigned int)ga;                        // global_addr[31:0]
  g0.w = ((unsigned int)(ga >> 32) & 0x01FFFFFFu) // global_addr[56:32]
         | (2u << 30);                            // type=2 ("image")
  v8i g1;
  g1[0] = (1 << 16);                              // workgroup_mask=0, data_size=1 (2B)
  g1[1] = 0;                                      // no atomic barrier
  g1[2] = tensor_w;                               // tensor_dim0
  g1[3] = tensor_h;                               // tensor_dim1
  g1[4] = (64 & 0xffff) | (16 << 16);             // tile_dim0=64, tile_dim1=16
  g1[5] = row_stride & 0xffff;                    // tensor_dim0_stride (lo)
  g1[6] = (row_stride >> 16);
  g1[7] = 0;
  v4i g2 = {0, 0, 0, 0};
  v4i g3 = {0, 0, 0, 0};
  v8i g4 = {0, 0, 0, 0, 0, 0, 0, 0};
  __builtin_amdgcn_tensor_load_to_lds(g0, g1, g2, g3, g4, 0);
  __builtin_amdgcn_s_wait_tensorcnt(0);
  __syncthreads();
  for (int i = threadIdx.x; i < 16 * 64; i += blockDim.x)
    dst[(long)blockIdx.x * 16 * 64 + i] = tile[i];
}
#endif

// ---------------- implicit-GEMM conv via WMMA ----------------

DEVINL _Float16 im2col_ld(const _Float16* __restrict__ X, long p, long NP,
                          int cin, int dy, int dx, int Cin, int H, int W,
                          int OH, int OW, int stride, int padH, int padW) {
  if (p >= NP) return (_Float16)0.f;
  int img = (int)(p / ((long)OH * OW));
  int rr  = (int)(p - (long)img * OH * OW);
  int oy = rr / OW, ox = rr - oy * OW;
  int iy = oy * stride + dy - padH;
  int ix = ox * stride + dx - padW;
  if (iy < 0 || iy >= H || ix < 0 || ix >= W) return (_Float16)0.f;
  return X[((long)(img * Cin + cin) * H + iy) * W + ix];
}

// LDS panel: column-major, Bs[buf][n * 72 + k], k in [0,64), stride 72 f16
// (144 B, 16B-aligned, padded to dodge bank conflicts). Each lane's B fragment
// is then two contiguous ds_load_b128 per WMMA.
#define BSTRIDE 72

__global__ __launch_bounds__(256) void k_conv_wmma(
    const _Float16* __restrict__ X,   // (Nb,Cin,H,W) f16 activations
    const _Float16* __restrict__ Wt,  // (Cout,Kp) f16 weights, Kp % 64 == 0
    const float* __restrict__ bias,   // Cout or nullptr
    float* __restrict__ Y,            // (Nb,Cout,OH,OW) f32
    int Nb, int Cin, int H, int W, int Cout, int KH, int KW,
    int OH, int OW, int stride, int padH, int padW, int Kp) {
  __shared__ _Float16 Bs[2][16 * BSTRIDE];

  const int tid = threadIdx.x, wave = tid >> 5, lane = tid & 31;
  const int col = lane & 15;
  const int ksel = (lane >> 4) ? 8 : 0;   // A/B fragment K sub-offset per lane half
  const long NP = (long)Nb * OH * OW;
  const long ptile = (long)blockIdx.x * 16;
  const int mbase = blockIdx.y * 128 + wave * 16;
  const int K = Cin * KH * KW, khw = KH * KW;

  // B-panel staging role: 4 f16 per thread (1024 total), same k, 4 consecutive n.
  const int se = tid * 4, sk = se >> 4, sn0 = se & 15;

  v8f acc = {};

  int mclamp = mbase + col;
  if (mclamp >= Cout) mclamp = Cout - 1;
  const _Float16* arow = Wt + (long)mclamp * Kp;

  for (int k0 = 0; k0 < K; k0 += 64) {
    const int buf = (k0 >> 6) & 1;
    // ---- cooperative im2col stage into LDS (column-major) ----
    {
      _Float16 vv[4];
      vv[0] = vv[1] = vv[2] = vv[3] = (_Float16)0.f;
      const int kg = k0 + sk;
      if (kg < K) {
        const int cin = kg / khw, r = kg - cin * khw;
        const int dy = r / KW, dx = r - dy * KW;
#pragma unroll
        for (int j = 0; j < 4; j++)
          vv[j] = im2col_ld(X, ptile + sn0 + j, NP, cin, dy, dx,
                            Cin, H, W, OH, OW, stride, padH, padW);
      }
#pragma unroll
      for (int j = 0; j < 4; j++)
        Bs[buf][(sn0 + j) * BSTRIDE + sk] = vv[j];
    }
    __syncthreads();

    // ---- A fragments: 16x64 f16 weights, four aligned b128 loads per lane ----
    const _Float16* ap = arow + k0 + ksel;
    __builtin_prefetch(ap + 64, 0, 0);          // global_prefetch next K tile
    v8h a0lo = *(const v8h*)(ap);               // K = ksel .. ksel+7
    v8h a0hi = *(const v8h*)(ap + 16);          // K = ksel+16 .. ksel+23
    v8h a1lo = *(const v8h*)(ap + 32);
    v8h a1hi = *(const v8h*)(ap + 48);

    // ---- B fragments: contiguous b128 reads from the stage column ----
    const _Float16* bc = &Bs[buf][col * BSTRIDE] + ksel;
    v8h b0lo = *(const v8h*)(bc);
    v8h b0hi = *(const v8h*)(bc + 16);
    v8h b1lo = *(const v8h*)(bc + 32);
    v8h b1hi = *(const v8h*)(bc + 48);

    v16h a0, a1, b0, b1;
#pragma unroll
    for (int i = 0; i < 8; i++) {
      a0[i] = a0lo[i]; a0[i + 8] = a0hi[i];
      a1[i] = a1lo[i]; a1[i + 8] = a1hi[i];
      b0[i] = b0lo[i]; b0[i + 8] = b0hi[i];
      b1[i] = b1lo[i]; b1[i + 8] = b1hi[i];
    }

    acc = __builtin_amdgcn_wmma_f32_16x16x32_f16(false, a0, false, b0,
                                                 (short)0, acc, false, false);
    acc = __builtin_amdgcn_wmma_f32_16x16x32_f16(false, a1, false, b1,
                                                 (short)0, acc, false, false);
  }

  // ---- epilogue: C layout VGPR i -> (M = i or 8+i, N = col) ----
  const long p = ptile + col;
  if (p < NP) {
    const int img = (int)(p / ((long)OH * OW));
    const int rr  = (int)(p - (long)img * OH * OW);
    const int oy = rr / OW, ox = rr - oy * OW;
    const int m0 = mbase + ((lane >> 4) ? 8 : 0);
#pragma unroll
    for (int i = 0; i < 8; i++) {
      const int m = m0 + i;
      if (m < Cout) {
        float v = acc[i] + (bias ? bias[m] : 0.f);
        Y[(((long)img * Cout + m) * OH + oy) * OW + ox] = v;
      }
    }
  }
}

// ---------------- batch-norm (training stats) ----------------

__global__ void k_bn_reduce(const float* __restrict__ x, float* __restrict__ sum,
                            float* __restrict__ ssum, int C, long cnt, long HW) {
  __shared__ float s1[256], s2[256];
  const int c = blockIdx.x, tid = threadIdx.x;
  float a = 0.f, b = 0.f;
  for (long i = tid + (long)blockIdx.y * 256; i < cnt; i += (long)gridDim.y * 256) {
    long img = i / HW, j = i - img * HW;
    float v = x[(img * C + c) * HW + j];
    a += v; b += v * v;
  }
  s1[tid] = a; s2[tid] = b;
  __syncthreads();
  for (int s = 128; s > 0; s >>= 1) {
    if (tid < s) { s1[tid] += s1[tid + s]; s2[tid] += s2[tid + s]; }
    __syncthreads();
  }
  if (tid == 0) { atomicAdd(&sum[c], s1[0]); atomicAdd(&ssum[c], s2[0]); }
}

__global__ void k_bn_finalize(const float* __restrict__ sum, const float* __restrict__ ssum,
                              const float* __restrict__ g, const float* __restrict__ be,
                              float* __restrict__ scale, float* __restrict__ shift,
                              int C, float inv_cnt) {
  int c = blockIdx.x * blockDim.x + threadIdx.x;
  if (c >= C) return;
  float m = sum[c] * inv_cnt;
  float v = ssum[c] * inv_cnt - m * m;
  float s = g[c] * rsqrtf(v + 1e-5f);
  scale[c] = s;
  shift[c] = be[c] - m * s;
}

__global__ void k_bn_apply(const float* __restrict__ x, const float* __restrict__ scale,
                           const float* __restrict__ shift, const float* __restrict__ resid,
                           int relu, _Float16* __restrict__ o16, float* __restrict__ o32,
                           int C, long HW, long n) {
  long i = (long)blockIdx.x * blockDim.x + threadIdx.x;
  if (i >= n) return;
  int c = (int)((i / HW) % C);
  float v = x[i] * scale[c] + shift[c];
  if (resid) v += resid[i];
  if (relu) v = fmaxf(v, 0.f);
  o16[i] = (_Float16)v;
  if (o32) o32[i] = v;
}

// ---------------- maxpool 3x3 s2 pad1 ----------------

__global__ void k_maxpool3s2(const _Float16* __restrict__ x, _Float16* __restrict__ o16,
                             float* __restrict__ o32, int NC, int H, int W, int OH, int OW) {
  long i = (long)blockIdx.x * blockDim.x + threadIdx.x;
  long n = (long)NC * OH * OW;
  if (i >= n) return;
  int ox = (int)(i % OW);
  long t = i / OW;
  int oy = (int)(t % OH);
  int nc = (int)(t / OH);
  float m = -3.0e38f;
  for (int dy = 0; dy < 3; dy++) {
    int iy = oy * 2 + dy - 1;
    if (iy < 0 || iy >= H) continue;
    for (int dx = 0; dx < 3; dx++) {
      int ix = ox * 2 + dx - 1;
      if (ix < 0 || ix >= W) continue;
      m = fmaxf(m, (float)x[((long)nc * H + iy) * W + ix]);
    }
  }
  o16[i] = (_Float16)m;
  o32[i] = m;
}

// ---------------- transposed conv (lhs_dilation=2) ----------------

__global__ void k_deconv(const _Float16* __restrict__ X, const _Float16* __restrict__ Wt,
                         const float* __restrict__ bias, float* __restrict__ Y,
                         int Nb, int Ci, int H, int W, int Co, int KH, int KW,
                         int OH, int OW, int padLoH, int padLoW) {
  long i = (long)blockIdx.x * blockDim.x + threadIdx.x;
  long n = (long)Nb * Co * OH * OW;
  if (i >= n) return;
  int ox = (int)(i % OW);
  long t = i / OW;
  int oy = (int)(t % OH); t /= OH;
  int co = (int)(t % Co);
  int img = (int)(t / Co);
  float acc = bias ? bias[co] : 0.f;
  for (int ky = 0; ky < KH; ky++) {
    int ty = oy + ky - padLoH;
    if (ty < 0 || (ty & 1)) continue;
    int iy = ty >> 1;
    if (iy >= H) continue;
    for (int kx = 0; kx < KW; kx++) {
      int tx = ox + kx - padLoW;
      if (tx < 0 || (tx & 1)) continue;
      int ix = tx >> 1;
      if (ix >= W) continue;
      const _Float16* xp = X + ((long)(img * Ci) * H + iy) * W + ix;
      const _Float16* wp = Wt + ((long)(co * Ci) * KH + ky) * KW + kx;
      for (int ci = 0; ci < Ci; ci++)
        acc += (float)xp[(long)ci * H * W] * (float)wp[(long)ci * KH * KW];
    }
  }
  Y[i] = acc;
}

// ---------------- message-passing scan step ----------------

__global__ void k_mp_step(float* __restrict__ e4, const float* __restrict__ w,
                          int Nb, int C, int HWtot, int sliceOff, int prevOff,
                          int jStride, int J) {
  int t = blockIdx.x * blockDim.x + threadIdx.x;
  int total = Nb * C * J;
  if (t >= total) return;
  int j = t % J;
  int rc = t / J;
  int co = rc % C;
  int n = rc / C;
  float acc = 0.f;
  for (int ci = 0; ci < C; ci++) {
    const float* pw = w + ((long)co * C + ci) * 9;
    const float* pc = e4 + ((long)n * C + ci) * HWtot + prevOff;
#pragma unroll
    for (int tt = 0; tt < 9; tt++) {
      int jj = j + tt - 4;
      if (jj >= 0 && jj < J) acc += pw[tt] * pc[(long)jj * jStride];
    }
  }
  float* po = e4 + ((long)n * C + co) * HWtot + sliceOff;
  po[(long)j * jStride] += fmaxf(acc, 0.f);
}

// ---------------- host orchestration ----------------

extern "C" void kernel_launch(void* const* d_in, const int* in_sizes, int n_in,
                              void* d_out, int out_size, void* d_ws, size_t ws_size,
                              hipStream_t stream) {
  (void)in_sizes; (void)n_in; (void)out_size; (void)ws_size;
  int idx = 0;
  auto F = [&]() -> const float* { return (const float*)d_in[idx++]; };

  const float* Xin = F();
  const float* w0 = F(); const float* g0 = F(); const float* be0 = F();
  struct Blk { const float *w1,*b1,*g1,*be1,*w2,*g2,*be2,*ws,*gs,*bes; };
  auto RB = [&](bool sc) {
    Blk b{}; b.w1=F(); b.b1=F(); b.g1=F(); b.be1=F(); b.w2=F(); b.g2=F(); b.be2=F();
    if (sc) { b.ws=F(); b.gs=F(); b.bes=F(); } else { b.ws=nullptr; b.gs=nullptr; b.bes=nullptr; }
    return b;
  };
  Blk L1[3] = {RB(true), RB(false), RB(false)};
  Blk L2[4] = {RB(true), RB(false), RB(false), RB(false)};
  Blk L3[6] = {RB(true), RB(false), RB(false), RB(false), RB(false), RB(false)};
  Blk L4[3] = {RB(true), RB(false), RB(false)};
  const float* mpw[4] = {F(), F(), F(), F()};
  struct Dec { const float *w1,*b1,*g1,*be1,*wd,*bd,*g2,*be2,*w3,*b3,*g3,*be3; };
  auto RD = [&]() {
    Dec d; d.w1=F(); d.b1=F(); d.g1=F(); d.be1=F(); d.wd=F(); d.bd=F(); d.g2=F(); d.be2=F();
    d.w3=F(); d.b3=F(); d.g3=F(); d.be3=F(); return d;
  };
  Dec D4 = RD(), D3 = RD(), D2 = RD(), D1 = RD();
  const float* wf1 = F(); const float* bf1 = F();
  const float* wf2 = F(); const float* bf2 = F();
  const float* wf3 = F(); const float* bf3 = F();

  const int Nb = 2;
  size_t off = 0;
  auto WS = [&](size_t bytes) -> void* {
    void* p = (char*)d_ws + off;
    off += (bytes + 255) & ~(size_t)255;
    return p;
  };
  _Float16* x16  = (_Float16*)WS(2L * 512 * 512 * sizeof(_Float16));
  _Float16* wbuf = (_Float16*)WS(16UL << 20);
  float* ssum = (float*)WS(512 * 4); float* ssq = (float*)WS(512 * 4);
  float* sscale = (float*)WS(512 * 4); float* sshift = (float*)WS(512 * 4);
  float* f32a = (float*)WS(68UL << 20);
  float* f32b = (float*)WS(68UL << 20);
  float* f32c = (float*)WS(36UL << 20);
  _Float16* h16a = (_Float16*)WS(34UL << 20);
  _Float16* h16b = (_Float16*)WS(34UL << 20);
  _Float16* h16c = (_Float16*)WS(18UL << 20);
  float* r32a = (float*)WS(9UL << 20);  float* r32b = (float*)WS(9UL << 20);
  _Float16* r16a = (_Float16*)WS(5UL << 20); _Float16* r16b = (_Float16*)WS(5UL << 20);
  float* e1f = (float*)WS(2L*64*128*128*4); _Float16* e1h = (_Float16*)WS(2L*64*128*128*2);
  float* e2f = (float*)WS(2L*128*64*64*4);  _Float16* e2h = (_Float16*)WS(2L*128*64*64*2);
  float* e3f = (float*)WS(2L*256*32*32*4);  _Float16* e3h = (_Float16*)WS(2L*256*32*32*2);
  float* e4f = (float*)WS(2L*512*16*16*4);  _Float16* e4h = (_Float16*)WS(2L*512*16*16*2);

  auto G1 = [&](long n) { return dim3((unsigned)((n + 255) / 256)); };

  auto conv = [&](const _Float16* Xi, int Cin, int H, int W, const float* w, const float* bias,
                  int Cout, int KH, int KW, int stride, int padH, int padW,
                  int OH, int OW, float* Y) {
    int K = Cin * KH * KW, Kp = (K + 63) & ~63;
    k_cvt_w<<<G1((long)Cout * Kp), 256, 0, stream>>>(w, wbuf, Cout, K, Kp);
    long NP = (long)Nb * OH * OW;
    dim3 grid((unsigned)((NP + 15) / 16), (unsigned)((Cout + 127) / 128));
    k_conv_wmma<<<grid, 256, 0, stream>>>(Xi, wbuf, bias, Y, Nb, Cin, H, W,
                                          Cout, KH, KW, OH, OW, stride, padH, padW, Kp);
  };
  auto bnstats = [&](const float* Y, int C, long HW, const float* g, const float* be) {
    k_zero<<<G1(C), 256, 0, stream>>>(ssum, C);
    k_zero<<<G1(C), 256, 0, stream>>>(ssq, C);
    long cnt = (long)Nb * HW;
    dim3 rg((unsigned)C, 16);
    k_bn_reduce<<<rg, 256, 0, stream>>>(Y, ssum, ssq, C, cnt, HW);
    k_bn_finalize<<<G1(C), 256, 0, stream>>>(ssum, ssq, g, be, sscale, sshift, C, 1.f / (float)cnt);
  };
  auto bnapply = [&](const float* Y, int C, long HW, const float* resid, int relu,
                     _Float16* o16, float* o32) {
    long n = (long)Nb * C * HW;
    k_bn_apply<<<G1(n), 256, 0, stream>>>(Y, sscale, sshift, resid, relu, o16, o32, C, HW, n);
  };

  auto resblock = [&](const _Float16* xin16, const float* xin32, int Cin, int H, int W,
                      const Blk& B, int stride, int Cout, _Float16* o16, float* o32) {
    int OH = (stride == 2) ? H / 2 : H, OW = (stride == 2) ? W / 2 : W;
    long HW = (long)OH * OW;
    conv(xin16, Cin, H, W, B.w1, B.b1, Cout, 3, 3, stride, 1, 1, OH, OW, f32a);
    bnstats(f32a, Cout, HW, B.g1, B.be1);
    bnapply(f32a, Cout, HW, nullptr, 1, h16a, nullptr);
    conv(h16a, Cout, OH, OW, B.w2, nullptr, Cout, 3, 3, 1, 1, 1, OH, OW, f32b);
    const float* sc32;
    if (B.ws) {
      conv(xin16, Cin, H, W, B.ws, nullptr, Cout, 1, 1, stride, 0, 0, OH, OW, f32c);
      bnstats(f32c, Cout, HW, B.gs, B.bes);
      bnapply(f32c, Cout, HW, nullptr, 0, h16c, f32c);
      sc32 = f32c;
    } else {
      sc32 = xin32;
    }
    bnstats(f32b, Cout, HW, B.g2, B.be2);
    bnapply(f32b, Cout, HW, sc32, 1, o16, o32);
  };

  auto run_layer = [&](const _Float16*& cur16, const float*& cur32, int& C, int& H, int& W,
                       const Blk* blks, int nb, int stride, int Cout,
                       _Float16* fin16, float* fin32) {
    for (int i = 0; i < nb; i++) {
      int s = (i == 0) ? stride : 1;
      _Float16* o16 = (i == nb - 1) ? fin16 : ((i & 1) ? r16b : r16a);
      float* o32 = (i == nb - 1) ? fin32 : ((i & 1) ? r32b : r32a);
      resblock(cur16, cur32, C, H, W, blks[i], s, Cout, o16, o32);
      cur16 = o16; cur32 = o32; C = Cout;
      if (s == 2) { H /= 2; W /= 2; }
    }
  };

  auto decoder = [&](const _Float16* in16, int Cin, int H, int W, const Dec& D,
                     int mid, int nf, _Float16* o16, float* o32) {
    long HW = (long)H * W;
    conv(in16, Cin, H, W, D.w1, D.b1, mid, 1, 1, 1, 0, 0, H, W, f32a);
    bnstats(f32a, mid, HW, D.g1, D.be1);
    bnapply(f32a, mid, HW, nullptr, 1, h16a, nullptr);
    int OH = 2 * H, OW = 2 * W;
    long OHW = (long)OH * OW;
    k_cvt16<<<G1((long)mid * mid * 9), 256, 0, stream>>>(D.wd, wbuf, (long)mid * mid * 9);
    k_deconv<<<G1((long)Nb * mid * OHW), 256, 0, stream>>>(h16a, wbuf, D.bd, f32b,
                                                           Nb, mid, H, W, mid, 3, 3, OH, OW, 1, 1);
    bnstats(f32b, mid, OHW, D.g2, D.be2);
    bnapply(f32b, mid, OHW, nullptr, 1, h16a, nullptr);
    conv(h16a, mid, OH, OW, D.w3, D.b3, nf, 1, 1, 1, 0, 0, OH, OW, f32a);
    bnstats(f32a, nf, OHW, D.g3, D.be3);
    bnapply(f32a, nf, OHW, nullptr, 1, o16, o32);
  };

  // ---------------- stem ----------------
  k_cvt16<<<G1(2L * 512 * 512), 256, 0, stream>>>(Xin, x16, 2L * 512 * 512);
  conv(x16, 1, 512, 512, w0, nullptr, 64, 7, 7, 2, 3, 3, 256, 256, f32a);
  bnstats(f32a, 64, 256L * 256, g0, be0);
  bnapply(f32a, 64, 256L * 256, nullptr, 1, h16a, nullptr);
  k_maxpool3s2<<<G1(2L * 64 * 128 * 128), 256, 0, stream>>>(h16a, r16b, r32b, 128, 256, 256, 128, 128);

  // ---------------- encoder ----------------
  const _Float16* cur16 = r16b; const float* cur32 = r32b;
  int C = 64, H = 128, W = 128;
  run_layer(cur16, cur32, C, H, W, L1, 3, 1, 64, e1h, e1f);
  run_layer(cur16, cur32, C, H, W, L2, 4, 2, 128, e2h, e2f);
  run_layer(cur16, cur32, C, H, W, L3, 6, 2, 256, e3h, e3f);
  run_layer(cur16, cur32, C, H, W, L4, 3, 2, 512, e4h, e4f);

  // ---------------- message passing on e4 (2,512,16,16) f32, in place ----------------
  {
    const int Cc = 512, Hh = 16, Ww = 16, HWtot = Hh * Ww;
    dim3 g = G1((long)Nb * Cc * 16);
    for (int i = 1; i < Hh; i++)          // down
      k_mp_step<<<g, 256, 0, stream>>>(e4f, mpw[0], Nb, Cc, HWtot, i * Ww, (i - 1) * Ww, 1, Ww);
    for (int i = Hh - 2; i >= 0; i--)     // up
      k_mp_step<<<g, 256, 0, stream>>>(e4f, mpw[1], Nb, Cc, HWtot, i * Ww, (i + 1) * Ww, 1, Ww);
    for (int i = 1; i < Ww; i++)          // right
      k_mp_step<<<g, 256, 0, stream>>>(e4f, mpw[2], Nb, Cc, HWtot, i, i - 1, Ww, Hh);
    for (int i = Ww - 2; i >= 0; i--)     // left
      k_mp_step<<<g, 256, 0, stream>>>(e4f, mpw[3], Nb, Cc, HWtot, i, i + 1, Ww, Hh);
    k_cvt16<<<G1(2L * 512 * 256), 256, 0, stream>>>(e4f, e4h, 2L * 512 * 256);
  }

  // ---------------- decoders + skips ----------------
  decoder(e4h, 512, 16, 16, D4, 128, 256, r16a, r32a);
  k_add_cvt<<<G1(2L * 256 * 32 * 32), 256, 0, stream>>>(r32a, e3f, r16a, 2L * 256 * 32 * 32);
  decoder(r16a, 256, 32, 32, D3, 64, 128, r16b, r32b);
  k_add_cvt<<<G1(2L * 128 * 64 * 64), 256, 0, stream>>>(r32b, e2f, r16b, 2L * 128 * 64 * 64);
  decoder(r16b, 128, 64, 64, D2, 32, 64, r16a, r32a);
  k_add_cvt<<<G1(2L * 64 * 128 * 128), 256, 0, stream>>>(r32a, e1f, r16a, 2L * 64 * 128 * 128);
  decoder(r16a, 64, 128, 128, D1, 16, 64, h16b, f32c);

  // ---------------- head ----------------
  long nfin = 2L * 32 * 512 * 512;
  k_cvt16<<<G1(32L * 64 * 16), 256, 0, stream>>>(wf1, wbuf, 32L * 64 * 16);
  k_deconv<<<G1(nfin), 256, 0, stream>>>(h16b, wbuf, bf1, f32a, Nb, 64, 256, 256, 32, 4, 4, 512, 512, 2, 2);
  k_relu_cvt<<<G1(nfin), 256, 0, stream>>>(f32a, h16a, nfin);
  conv(h16a, 32, 512, 512, wf2, bf2, 32, 3, 3, 1, 1, 1, 512, 512, f32b);
  k_relu_cvt<<<G1(nfin), 256, 0, stream>>>(f32b, h16b, nfin);
  conv(h16b, 32, 512, 512, wf3, bf3, 1, 3, 3, 1, 1, 1, 512, 512, f32a);
  k_sigmoid<<<G1(2L * 512 * 512), 256, 0, stream>>>(f32a, (float*)d_out, 2L * 512 * 512);
}
